// TersoffBlock_12128987644522
// MI455X (gfx1250) — matrix operand order repaired and checked by reading, compiled, not run on hardware
//
#include <hip/hip_runtime.h>
#include <hip/hip_bf16.h>
#include <stddef.h>

// ---------------------------------------------------------------------------
// TersoffBlock on MI455X (gfx1250, wave32, WMMA).
// All dense GEMMs use v_wmma_f32_16x16x32_bf16 (bf16 in, fp32 accumulate).
// Per K-step, ALL B fragments are loaded up-front into distinct registers and
// the wmma ops issue back-to-back afterwards -> deep load/wmma overlap instead
// of a full s_wait_loadcnt drain per wmma.
// Triplet stage: one wave per triplet (coalesced L2 gathers of xij rows).
// Segment softmax via order-preserving uint atomicMax + float atomicAdd.
// FFN: fused Linear->SiLU->Linear, hidden tile in 132KB static LDS.
// ---------------------------------------------------------------------------

#define Nn 10000
#define Ee 160000
#define Tt 1600000
#define Dd 256
#define Cc 64
#define Hh 1024

typedef __attribute__((ext_vector_type(16))) __bf16 v16bf;
typedef __attribute__((ext_vector_type(8)))  __bf16 v8bf;
typedef __attribute__((ext_vector_type(8)))  float  v8f;

__device__ __forceinline__ float silu_f(float v) { return v / (1.0f + __expf(-v)); }

__device__ __forceinline__ __bf16 f2bf(float f) {
    unsigned u = __float_as_uint(f);
    unsigned r = u + 0x7FFFu + ((u >> 16) & 1u);   // round-to-nearest-even
    unsigned short h = (unsigned short)(r >> 16);
    return __builtin_bit_cast(__bf16, h);
}

// order-preserving float <-> uint key for atomicMax-based segment max
__device__ __forceinline__ unsigned fkey(float f) {
    unsigned u = __float_as_uint(f);
    return (u & 0x80000000u) ? ~u : (u | 0x80000000u);
}
__device__ __forceinline__ float funkey(unsigned k) {
    unsigned u = (k & 0x80000000u) ? (k & 0x7FFFFFFFu) : ~k;
    return __uint_as_float(u);
}

// ---- WMMA fragment helpers (contiguous k order, same for A and B) ------------------
// A: 16x32 bf16, row-major source. lane m=lane&15 is row; h=lane>>4.
// elements 0..7  -> K = 16h + i        (VGPR0-3)
// elements 8..15 -> K = 16h + i       (VGPR4-7)
__device__ __forceinline__ v16bf load_frag_a(const __bf16* A, int lda, int row0, int k0) {
    int lane = threadIdx.x & 31;
    int m = lane & 15, h = lane >> 4;
    const __bf16* base = A + (size_t)(row0 + m) * lda + k0;
    v8bf lo = *(const v8bf*)(base + 16 * h);
    v8bf hi = *(const v8bf*)(base + 16 * h + 8);
    v16bf f;
#pragma unroll
    for (int i = 0; i < 8; ++i) { f[i] = lo[i]; f[8 + i] = hi[i]; }
    return f;
}

// B: 32x16 bf16 tile taken from a PRE-TRANSPOSED weight Bt[N][K] (ldk = K).
// lane n=lane&15 is column; element i holds K = 16h + i -> contiguous 32B.
__device__ __forceinline__ v16bf load_frag_bT(const __bf16* Bt, int ldk, int k0, int n0) {
    int lane = threadIdx.x & 31;
    int n = lane & 15, h = lane >> 4;
    const __bf16* base = Bt + (size_t)(n0 + n) * ldk + k0 + 16 * h;
    v8bf lo = *(const v8bf*)(base);
    v8bf hi = *(const v8bf*)(base + 8);
    v16bf f;
#pragma unroll
    for (int i = 0; i < 8; ++i) { f[i] = lo[i]; f[8 + i] = hi[i]; }
    return f;
}

__device__ __forceinline__ v8f wmma_bf16(v16bf a, v16bf b, v8f c) {
    return __builtin_amdgcn_wmma_f32_16x16x32_bf16(
        false, a, false, b, (short)0, c, false, false);
}

// ---------------------------------------------------------------------------
// Conversion / prep kernels
// ---------------------------------------------------------------------------
__global__ void cvt_bf16_kernel(const float* __restrict__ in, __bf16* __restrict__ out, int n) {
    int i = blockIdx.x * blockDim.x + threadIdx.x;
    if (i < n) out[i] = f2bf(in[i]);
}

// in: row-major (K x N) fp32  ->  out: bf16 [N][K] (transposed, fragment-friendly)
__global__ void cvtT_bf16_kernel(const float* __restrict__ in, __bf16* __restrict__ out, int K, int N) {
    int i = blockIdx.x * blockDim.x + threadIdx.x;
    if (i < K * N) {
        int k = i / N, n = i % N;
        out[(size_t)n * K + k] = f2bf(in[i]);
    }
}

__global__ void rnorm_kernel(const float* __restrict__ r, float* __restrict__ rn, int E) {
    int e = blockIdx.x * blockDim.x + threadIdx.x;
    if (e >= E) return;
    float x = r[3 * e], y = r[3 * e + 1], z = r[3 * e + 2];
    float inv = -rsqrtf(x * x + y * y + z * z);   // rnorm = -r / |r|
    rn[3 * e] = x * inv; rn[3 * e + 1] = y * inv; rn[3 * e + 2] = z * inv;
}

__global__ void init_kernel(float* __restrict__ ft, unsigned* __restrict__ segmax,
                            float* __restrict__ den, int E, int EC) {
    int i = blockIdx.x * blockDim.x + threadIdx.x;
    if (i < EC) ft[i] = 0.0f;
    if (i < E) { segmax[i] = 0u; den[i] = 0.0f; }   // 0u < fkey(any finite) -> acts as -inf
}

// ---------------------------------------------------------------------------
// WMMA GEMM: out(M x 64) = A(M x K)bf16 @ Bt(64 x K)bf16 + bias, with optional
// per-row gather-add epilogue (xij = y@We + b + xs[src] + xd[dst]).
// One wave owns a 16x64 strip. Per K-step: 1 A frag + 4 B frags loaded into
// distinct registers, then 4 back-to-back wmma into 4 resident accumulators.
// ---------------------------------------------------------------------------
__global__ void gemm_bias_gather_kernel(
    const __bf16* __restrict__ A, int lda,
    const __bf16* __restrict__ Bt, int K,
    const float* __restrict__ bias,
    const int* __restrict__ gsrc, const float* __restrict__ addS,
    const int* __restrict__ gdst, const float* __restrict__ addD,
    float* __restrict__ outF, int ldo, int Mtiles)
{
    int w = blockIdx.x * (blockDim.x >> 5) + (threadIdx.x >> 5);
    if (w >= Mtiles) return;
    int row0 = w * 16;

    v8f acc0 = {}, acc1 = {}, acc2 = {}, acc3 = {};
    for (int k0 = 0; k0 < K; k0 += 32) {
        v16bf a  = load_frag_a(A, lda, row0, k0);
        v16bf b0 = load_frag_bT(Bt, K, k0, 0);
        v16bf b1 = load_frag_bT(Bt, K, k0, 16);
        v16bf b2 = load_frag_bT(Bt, K, k0, 32);
        v16bf b3 = load_frag_bT(Bt, K, k0, 48);
        acc0 = wmma_bf16(a, b0, acc0);
        acc1 = wmma_bf16(a, b1, acc1);
        acc2 = wmma_bf16(a, b2, acc2);
        acc3 = wmma_bf16(a, b3, acc3);
    }

    int lane = threadIdx.x & 31;
    int n = lane & 15, h = lane >> 4;
    float vout[4][8];
#pragma unroll
    for (int rr = 0; rr < 8; ++rr) {
        vout[0][rr] = acc0[rr]; vout[1][rr] = acc1[rr];
        vout[2][rr] = acc2[rr]; vout[3][rr] = acc3[rr];
    }
#pragma unroll
    for (int rr = 0; rr < 8; ++rr) {
        int row = row0 + rr + 8 * h;               // C layout: VGPR rr -> M = rr + 8h
        const float* aS = nullptr; const float* aD = nullptr;
        if (gsrc) {
            aS = addS + (size_t)gsrc[row] * Cc;
            aD = addD + (size_t)gdst[row] * Cc;
        }
#pragma unroll
        for (int nt = 0; nt < 4; ++nt) {
            int col = nt * 16 + n;
            float v = vout[nt][rr] + bias[col];
            if (gsrc) v += aS[col] + aD[col];
            outF[(size_t)row * ldo + col] = v;
        }
    }
}

// ---------------------------------------------------------------------------
// Triplet logits: one wave per triplet. lane l covers channels 2l, 2l+1.
// z_n = cos(n * acos(cos_jik)); e = silu(z + xij[ts] + xij[td]); logit = e.attn
// ---------------------------------------------------------------------------
__global__ void logits_kernel(const float* __restrict__ rn,
                              const int* __restrict__ tsrc, const int* __restrict__ tdst,
                              const float* __restrict__ xij, const float* __restrict__ attn,
                              float* __restrict__ logits, unsigned* __restrict__ segmax, int T)
{
    int t = blockIdx.x * (blockDim.x >> 5) + (threadIdx.x >> 5);
    if (t >= T) return;
    int lane = threadIdx.x & 31;
    int ts = tsrc[t], td = tdst[t];

    float ax = rn[3 * ts], ay = rn[3 * ts + 1], az = rn[3 * ts + 2];
    float bx = rn[3 * td], by = rn[3 * td + 1], bz = rn[3 * td + 2];
    float c = ax * bx + ay * by + az * bz;
    c = fminf(fmaxf(c, -1.0f + 1e-6f), 1.0f - 1e-6f);
    float theta = acosf(c);

    int c0 = lane * 2;
    float2 xs2 = *(const float2*)(xij + (size_t)ts * Cc + c0);
    float2 xd2 = *(const float2*)(xij + (size_t)td * Cc + c0);
    float2 at2 = *(const float2*)(attn + c0);

    float z0 = __cosf(theta * (float)c0);
    float z1 = __cosf(theta * (float)(c0 + 1));
    float e0 = silu_f(z0 + xs2.x + xd2.x);
    float e1 = silu_f(z1 + xs2.y + xd2.y);
    float p = e0 * at2.x + e1 * at2.y;
#pragma unroll
    for (int off = 16; off > 0; off >>= 1) p += __shfl_xor(p, off, 32);

    if (lane == 0) {
        logits[t] = p;
        atomicMax(&segmax[td], fkey(p));
    }
}

__global__ void expden_kernel(float* __restrict__ logits, const int* __restrict__ tdst,
                              const unsigned* __restrict__ segmax, float* __restrict__ den, int T)
{
    int t = blockIdx.x * blockDim.x + threadIdx.x;
    if (t >= T) return;
    int td = tdst[t];
    float ex = __expf(logits[t] - funkey(segmax[td]));
    logits[t] = ex;                                  // overwrite with exp value
    atomicAdd(&den[td], ex);
}

// ft[td] += xij[ts] * a   (one wave per triplet; 2 channels/lane; L2-resident table)
__global__ void scatter_kernel(const float* __restrict__ exv, const float* __restrict__ den,
                               const int* __restrict__ tsrc, const int* __restrict__ tdst,
                               const float* __restrict__ xij, float* __restrict__ ft, int T)
{
    int t = blockIdx.x * (blockDim.x >> 5) + (threadIdx.x >> 5);
    if (t >= T) return;
    int lane = threadIdx.x & 31;
    int ts = tsrc[t], td = tdst[t];
    float a = exv[t] / den[td];
    int c0 = lane * 2;
    float2 xv = *(const float2*)(xij + (size_t)ts * Cc + c0);
    atomicAdd(&ft[(size_t)td * Cc + c0],     xv.x * a);
    atomicAdd(&ft[(size_t)td * Cc + c0 + 1], xv.y * a);
}

// ---------------------------------------------------------------------------
// Fused FFN: out = silu(ft @ W1 + b1) @ W2 + b2.
// Block = 256 threads (8 waves) handles 64 edge rows; wave w owns M-tile w&3
// and N-half w>>2. Hidden (64 x 1024 bf16, ld padded to 1032 against LDS bank
// conflicts) lives in 132KB static LDS (<= 320KB/WGP).
// GEMM1: both A fragments (K=64) loaded ONCE per wave, reused by 32 N tiles.
// GEMM2: per K-step, 1 LDS A frag + 8 global B frags in distinct registers,
// then 8 back-to-back wmma into 8 resident accumulators.
// ---------------------------------------------------------------------------
#define HPAD 1032

__global__ void ffn_kernel(const __bf16* __restrict__ ftbf,
                           const __bf16* __restrict__ W1t, const float* __restrict__ b1,
                           const __bf16* __restrict__ W2t, const float* __restrict__ b2,
                           float* __restrict__ out)
{
    __shared__ __bf16 hid[64 * HPAD];
    int wave = threadIdx.x >> 5;
    int lane = threadIdx.x & 31;
    int n = lane & 15, h = lane >> 4;
    int row_base = blockIdx.x * 64;
    int mt   = wave & 3;       // M tile within the 64-row block
    int half = wave >> 2;      // N-range split (2 waves cover each M tile)

    // GEMM1: hidden = silu(ft(64x64) @ W1(64x1024) + b1) -> LDS bf16
    {
        int row0 = row_base + mt * 16;
        v16bf a0 = load_frag_a(ftbf, Cc, row0, 0);
        v16bf a1 = load_frag_a(ftbf, Cc, row0, 32);
        for (int nt = half * 32; nt < half * 32 + 32; nt += 2) {
            int n0 = nt * 16;
            v16bf b00 = load_frag_bT(W1t, Cc, 0,  n0);
            v16bf b01 = load_frag_bT(W1t, Cc, 32, n0);
            v16bf b10 = load_frag_bT(W1t, Cc, 0,  n0 + 16);
            v16bf b11 = load_frag_bT(W1t, Cc, 32, n0 + 16);
            v8f accA = {}, accB = {};
            accA = wmma_bf16(a0, b00, accA);
            accB = wmma_bf16(a0, b10, accB);
            accA = wmma_bf16(a1, b01, accA);
            accB = wmma_bf16(a1, b11, accB);
            float bvA = b1[n0 + n], bvB = b1[n0 + 16 + n];
#pragma unroll
            for (int rr = 0; rr < 8; ++rr) {
                size_t rowoff = (size_t)(mt * 16 + rr + 8 * h) * HPAD;
                hid[rowoff + n0 + n]      = f2bf(silu_f(accA[rr] + bvA));
                hid[rowoff + n0 + 16 + n] = f2bf(silu_f(accB[rr] + bvB));
            }
        }
    }
    __syncthreads();

    // GEMM2: out = hidden(64x1024) @ W2(1024x256) + b2
    {
        int m0 = mt * 16;
        int nb = half * 128;                       // 8 N tiles (128 cols) per wave
        v8f acc0 = {}, acc1 = {}, acc2 = {}, acc3 = {};
        v8f acc4 = {}, acc5 = {}, acc6 = {}, acc7 = {};
        for (int k0 = 0; k0 < Hh; k0 += 32) {
            v16bf a  = load_frag_a(hid, HPAD, m0, k0);
            v16bf b0 = load_frag_bT(W2t, Hh, k0, nb);
            v16bf b1f = load_frag_bT(W2t, Hh, k0, nb + 16);
            v16bf b2f = load_frag_bT(W2t, Hh, k0, nb + 32);
            v16bf b3 = load_frag_bT(W2t, Hh, k0, nb + 48);
            v16bf b4 = load_frag_bT(W2t, Hh, k0, nb + 64);
            v16bf b5 = load_frag_bT(W2t, Hh, k0, nb + 80);
            v16bf b6 = load_frag_bT(W2t, Hh, k0, nb + 96);
            v16bf b7 = load_frag_bT(W2t, Hh, k0, nb + 112);
            acc0 = wmma_bf16(a, b0, acc0);
            acc1 = wmma_bf16(a, b1f, acc1);
            acc2 = wmma_bf16(a, b2f, acc2);
            acc3 = wmma_bf16(a, b3, acc3);
            acc4 = wmma_bf16(a, b4, acc4);
            acc5 = wmma_bf16(a, b5, acc5);
            acc6 = wmma_bf16(a, b6, acc6);
            acc7 = wmma_bf16(a, b7, acc7);
        }
        float vout[8][8];
#pragma unroll
        for (int rr = 0; rr < 8; ++rr) {
            vout[0][rr] = acc0[rr]; vout[1][rr] = acc1[rr];
            vout[2][rr] = acc2[rr]; vout[3][rr] = acc3[rr];
            vout[4][rr] = acc4[rr]; vout[5][rr] = acc5[rr];
            vout[6][rr] = acc6[rr]; vout[7][rr] = acc7[rr];
        }
#pragma unroll
        for (int j = 0; j < 8; ++j) {
            int n0 = nb + j * 16;
            float bv = b2[n0 + n];
#pragma unroll
            for (int rr = 0; rr < 8; ++rr)
                out[(size_t)(row_base + m0 + rr + 8 * h) * Dd + n0 + n] = vout[j][rr] + bv;
        }
    }
}

// ---------------------------------------------------------------------------
extern "C" void kernel_launch(void* const* d_in, const int* in_sizes, int n_in,
                              void* d_out, int out_size, void* d_ws, size_t ws_size,
                              hipStream_t stream) {
    (void)in_sizes; (void)n_in; (void)out_size; (void)ws_size;

    const float* x      = (const float*)d_in[0];
    const float* y      = (const float*)d_in[1];
    const float* r      = (const float*)d_in[2];
    const int*   src    = (const int*)d_in[3];
    const int*   dst    = (const int*)d_in[4];
    const int*   tsrc   = (const int*)d_in[5];
    const int*   tdst   = (const int*)d_in[6];
    const float* W_src  = (const float*)d_in[7];
    const float* b_src  = (const float*)d_in[8];
    const float* W_dst  = (const float*)d_in[9];
    const float* b_dst  = (const float*)d_in[10];
    const float* W_edge = (const float*)d_in[11];
    const float* b_edge = (const float*)d_in[12];
    const float* attn   = (const float*)d_in[13];
    const float* W1     = (const float*)d_in[14];
    const float* b1     = (const float*)d_in[15];
    const float* W2     = (const float*)d_in[16];
    const float* b2     = (const float*)d_in[17];
    float* out = (float*)d_out;

    // ---- workspace carve-out (~205 MB total) ----
    char* base = (char*)d_ws; size_t off = 0;
    auto alloc = [&](size_t bytes) -> char* {
        char* q = base + off;
        off = (off + bytes + 255) & ~(size_t)255;
        return q;
    };
    __bf16* xbf    = (__bf16*)alloc((size_t)Nn * Dd * 2);
    __bf16* ybf    = (__bf16*)alloc((size_t)Ee * Dd * 2);
    __bf16* WsrcT  = (__bf16*)alloc((size_t)Dd * Cc * 2);
    __bf16* WdstT  = (__bf16*)alloc((size_t)Dd * Cc * 2);
    __bf16* WedgeT = (__bf16*)alloc((size_t)Dd * Cc * 2);
    __bf16* W1T    = (__bf16*)alloc((size_t)Cc * Hh * 2);
    __bf16* W2T    = (__bf16*)alloc((size_t)Hh * Dd * 2);
    float* xs      = (float*)alloc((size_t)Nn * Cc * 4);
    float* xd      = (float*)alloc((size_t)Nn * Cc * 4);
    float* xij     = (float*)alloc((size_t)Ee * Cc * 4);
    float* rnrm    = (float*)alloc((size_t)Ee * 3 * 4);
    float* logits  = (float*)alloc((size_t)Tt * 4);
    unsigned* smax = (unsigned*)alloc((size_t)Ee * 4);
    float* den     = (float*)alloc((size_t)Ee * 4);
    float* ft      = (float*)alloc((size_t)Ee * Cc * 4);
    __bf16* ftbf   = (__bf16*)alloc((size_t)Ee * Cc * 2);

    const int TPB = 256;

    // 1) precision conversion (bf16 activations, transposed bf16 weights)
    cvt_bf16_kernel<<<(Nn * Dd + TPB - 1) / TPB, TPB, 0, stream>>>(x, xbf, Nn * Dd);
    cvt_bf16_kernel<<<(Ee * Dd + TPB - 1) / TPB, TPB, 0, stream>>>(y, ybf, Ee * Dd);
    cvtT_bf16_kernel<<<(Dd * Cc + TPB - 1) / TPB, TPB, 0, stream>>>(W_src,  WsrcT,  Dd, Cc);
    cvtT_bf16_kernel<<<(Dd * Cc + TPB - 1) / TPB, TPB, 0, stream>>>(W_dst,  WdstT,  Dd, Cc);
    cvtT_bf16_kernel<<<(Dd * Cc + TPB - 1) / TPB, TPB, 0, stream>>>(W_edge, WedgeT, Dd, Cc);
    cvtT_bf16_kernel<<<(Cc * Hh + TPB - 1) / TPB, TPB, 0, stream>>>(W1, W1T, Cc, Hh);
    cvtT_bf16_kernel<<<(Hh * Dd + TPB - 1) / TPB, TPB, 0, stream>>>(W2, W2T, Hh, Dd);

    // 2) node projections xs = x@W_src+b, xd = x@W_dst+b  (WMMA, 16x64/wave)
    {
        int mtiles = Nn / 16;
        int blocks = (mtiles + 7) / 8;
        gemm_bias_gather_kernel<<<blocks, TPB, 0, stream>>>(
            xbf, Dd, WsrcT, Dd, b_src, nullptr, nullptr, nullptr, nullptr,
            xs, Cc, mtiles);
        gemm_bias_gather_kernel<<<blocks, TPB, 0, stream>>>(
            xbf, Dd, WdstT, Dd, b_dst, nullptr, nullptr, nullptr, nullptr,
            xd, Cc, mtiles);
    }

    // 3) xij = y@W_edge + b_edge + xs[src] + xd[dst]  (WMMA + gather epilogue)
    {
        int mtiles = Ee / 16;
        int blocks = (mtiles + 7) / 8;
        gemm_bias_gather_kernel<<<blocks, TPB, 0, stream>>>(
            ybf, Dd, WedgeT, Dd, b_edge, src, xs, dst, xd,
            xij, Cc, mtiles);
    }

    // 4) geometry + accumulator init
    rnorm_kernel<<<(Ee + TPB - 1) / TPB, TPB, 0, stream>>>(r, rnrm, Ee);
    init_kernel<<<(Ee * Cc + TPB - 1) / TPB, TPB, 0, stream>>>(ft, smax, den, Ee, Ee * Cc);

    // 5) triplet logits + segment max  (wave per triplet)
    logits_kernel<<<Tt / 8, TPB, 0, stream>>>(rnrm, tsrc, tdst, xij, attn, logits, smax, Tt);

    // 6) exp + segment sum
    expden_kernel<<<(Tt + TPB - 1) / TPB, TPB, 0, stream>>>(logits, tdst, smax, den, Tt);

    // 7) attention-weighted scatter into ft
    scatter_kernel<<<Tt / 8, TPB, 0, stream>>>(logits, den, tsrc, tdst, xij, ft, Tt);

    // 8) fused FFN (WMMA x2, hidden in LDS)
    cvt_bf16_kernel<<<(Ee * Cc + TPB - 1) / TPB, TPB, 0, stream>>>(ft, ftbf, Ee * Cc);
    ffn_kernel<<<Ee / 64, TPB, 0, stream>>>(ftbf, W1T, b1, W2T, b2, out);
}
